// UHGGINLayer_21328807592547
// MI455X (gfx1250) — matrix operand order repaired and checked by reading, not compiled, not run on hardware
//
#include <hip/hip_runtime.h>
#include <hip/hip_bf16.h>

// ---------------------------------------------------------------------------
// UHG-GIN layer for MI455X (gfx1250, wave32, WMMA).
//   aggregate (edge softmax + weighted scatter + L2 normalize)
//   -> combined = x + neigh
//   -> WMMA GEMM1 (f16 in, f32 acc) + bias + ReLU
//   -> WMMA GEMM2 + bias + LayerNorm
// Workspace intermediates are padded to a multiple of 128 rows so the GEMM
// bodies and GEMM1 epilogue run with no per-element guards (full EXEC).
// ---------------------------------------------------------------------------

typedef __attribute__((ext_vector_type(16))) _Float16 v16h;
typedef __attribute__((ext_vector_type(8)))  _Float16 v8h;
typedef __attribute__((ext_vector_type(8)))  float    v8f;

#define D 128   // feature dim (in == out == 128)

// ---- ordered-uint encoding of float for atomicMax-based segment max --------
static __device__ __forceinline__ unsigned f2ord(float f) {
  unsigned b = __float_as_uint(f);
  return (b & 0x80000000u) ? ~b : (b | 0x80000000u);
}
static __device__ __forceinline__ float ord2f(unsigned k) {
  unsigned b = (k & 0x80000000u) ? (k & 0x7FFFFFFFu) : ~k;
  return __uint_as_float(b);
}

// ---- K0: zero agg/denom, init segment-max to "-inf" (encoded 0) ------------
__global__ void k_init(float* __restrict__ agg, unsigned* __restrict__ mo,
                       float* __restrict__ dn, long long totalAgg, int n) {
  long long i = (long long)blockIdx.x * 256 + threadIdx.x;
  if (i < totalAgg) agg[i] = 0.f;
  if (i < n) { mo[i] = 0u; dn[i] = 0.f; }
}

// ---- K1: convert W1/W2 to f16, pre-swizzled into B-fragment layout ---------
// Fragment layout (ISA 7.12.2, 16-bit B 32x16, wave32):
//   lane n holds column N = (lane&15); lanes 0-15 hold K=kt*32+0..7,16..23,
//   lanes 16-31 hold K=kt*32+8..15,24..31; 16 halves contiguous per lane.
// Storage: Ws[((nt*4 + kt)*32 + lane)*16 + j]
__global__ void k_wswz(const float* __restrict__ W1, const float* __restrict__ W2,
                       _Float16* __restrict__ W1s, _Float16* __restrict__ W2s) {
  int tid = blockIdx.x * 256 + threadIdx.x;        // 0..32767
  const float* W  = (tid & 16384) ? W2  : W1;
  _Float16*    Ds = (tid & 16384) ? W2s : W1s;
  int r    = tid & 16383;
  int nt   = r >> 11;
  int kt   = (r >> 9) & 3;
  int lane = (r >> 4) & 31;
  int j    = r & 15;
  int n = nt * 16 + (lane & 15);
  int k = kt * 32 + ((lane >> 4) * 8) + (j & 7) + ((j >> 3) * 16);
  Ds[r] = (_Float16)W[k * D + n];                  // W is [K=128][N=128] row-major
}

// ---- K2: per-target segment max of edge weights ----------------------------
__global__ void k_emax(const int* __restrict__ ei, const float* __restrict__ w,
                       unsigned* __restrict__ mo, int E) {
  int e = blockIdx.x * 256 + threadIdx.x;
  if (e >= E) return;
  int t = ei[E + e];                               // edge_index row 1 = target
  atomicMax(&mo[t], f2ord(w[e]));
}

// ---- K3: stable softmax numerator + denominator ----------------------------
__global__ void k_eexp(const int* __restrict__ ei, const float* __restrict__ w,
                       const unsigned* __restrict__ mo, float* __restrict__ ew,
                       float* __restrict__ dn, int E) {
  int e = blockIdx.x * 256 + threadIdx.x;
  if (e >= E) return;
  int t = ei[E + e];
  float v = __expf(w[e] - ord2f(mo[t]));
  ew[e] = v;
  atomicAdd(&dn[t], v);
}

// ---- K4: one wave per edge: gather x[src] (float4/lane), scatter-add -------
__global__ void k_scatter(const int* __restrict__ ei, const float* __restrict__ ew,
                          const float* __restrict__ dn, const float* __restrict__ x,
                          float* __restrict__ agg, int E) {
  int wid  = (blockIdx.x * 256 + threadIdx.x) >> 5;   // one edge per wave32
  int lane = threadIdx.x & 31;
  if (wid >= E) return;
  int s = ei[wid];
  int t = ei[E + wid];
  float w = ew[wid] / dn[t];                          // per-node softmax weight
  const float4 xv = ((const float4*)(x + (long long)s * D))[lane];
  float* ap = agg + (long long)t * D + lane * 4;
  atomicAdd(ap + 0, w * xv.x);
  atomicAdd(ap + 1, w * xv.y);
  atomicAdd(ap + 2, w * xv.z);
  atomicAdd(ap + 3, w * xv.w);
}

// ---- K5: one wave per node: projective average + combine, emit f16 ---------
__global__ void k_combine(const float* __restrict__ agg, const float* __restrict__ x,
                          _Float16* __restrict__ cmb, int n) {
  int node = (blockIdx.x * 256 + threadIdx.x) >> 5;
  int lane = threadIdx.x & 31;
  if (node >= n) return;
  const float4 a = ((const float4*)(agg + (long long)node * D))[lane];
  float s = a.x * a.x + a.y * a.y + a.z * a.z + a.w * a.w;
#pragma unroll
  for (int m = 1; m < 32; m <<= 1) s += __shfl_xor(s, m, 32);
  float inv = 1.f / (sqrtf(s) + 1e-9f);               // zero rows stay zero
  const float4 xv = ((const float4*)(x + (long long)node * D))[lane];
  _Float16* cp = cmb + (long long)node * D + lane * 4;
  cp[0] = (_Float16)(xv.x + a.x * inv);
  cp[1] = (_Float16)(xv.y + a.y * inv);
  cp[2] = (_Float16)(xv.z + a.z * inv);
  cp[3] = (_Float16)(xv.w + a.w * inv);
}

// ---- A fragment loader: 16-bit A 16x32 layout (ISA 7.12.2) -----------------
// lane&15 = row M; lanes 0-15: K = koff 0..7 and 16..23; lanes 16-31: +8.
static __device__ __forceinline__ v16h load_afrag(const _Float16* __restrict__ A,
                                                  int row, int kt, int koff) {
  const _Float16* ap = A + (long long)row * D + kt * 32 + koff;
  v8h lo = *(const v8h*)ap;                           // global_load_b128
  v8h hi = *(const v8h*)(ap + 16);                    // global_load_b128
  v16h a;
#pragma unroll
  for (int j = 0; j < 8; j++) { a[j] = lo[j]; a[j + 8] = hi[j]; }
  return a;
}

// ---- K6: GEMM1 (combined @ W1 + b1, ReLU) via v_wmma_f32_16x16x32_f16 ------
// A and Out buffers are padded to a multiple of 128 rows: no guards anywhere.
// (Garbage in padded A rows only affects padded D rows: D row m <- A row m.)
__global__ __launch_bounds__(256, 1)
void k_gemm1(const _Float16* __restrict__ A, const _Float16* __restrict__ Wsw,
             const float* __restrict__ bias, _Float16* __restrict__ Out) {
  __shared__ _Float16 Ws[D * D];                      // 32 KB pre-swizzled W tile
  {
    const uint4* src = (const uint4*)Wsw;
    uint4* dst = (uint4*)Ws;
    for (int i = threadIdx.x; i < (D * D * 2) / 16; i += 256) dst[i] = src[i];
  }
  __syncthreads();
  int wave = threadIdx.x >> 5, lane = threadIdx.x & 31;
  int rowBase = blockIdx.x * 128 + wave * 16;         // 8 waves -> 128 rows/block
  int arow = rowBase + (lane & 15);
  int koff = (lane >> 4) * 8;

  v16h a[4];
#pragma unroll
  for (int kt = 0; kt < 4; kt++) a[kt] = load_afrag(A, arow, kt, koff);

  int colLo = lane & 15;
  int mBase = rowBase + ((lane >> 4) << 3);
#pragma unroll
  for (int nt = 0; nt < 8; nt++) {
    v8f acc = {};
#pragma unroll
    for (int kt = 0; kt < 4; kt++) {
      v16h b = *(const v16h*)&Ws[((nt * 4 + kt) * 32 + lane) * 16];
      acc = __builtin_amdgcn_wmma_f32_16x16x32_f16(false, a[kt], false, b,
                                                   (short)0, acc, false, false);
    }
    int col = nt * 16 + colLo;
    float bb = bias[col];
#pragma unroll
    for (int r = 0; r < 8; r++) {
      float v = acc[r] + bb;
      Out[(long long)(mBase + r) * D + col] = (_Float16)(v > 0.f ? v : 0.f);
    }
  }
}

// ---- K7: GEMM2 (h1 @ W2 + b2) fused with LayerNorm -> f32 out --------------
// A (h1) is padded; output guard is wave-uniform: only tail waves of the last
// block take the per-element guarded path.
__global__ __launch_bounds__(256, 1)
void k_gemm2(const _Float16* __restrict__ A, const _Float16* __restrict__ Wsw,
             const float* __restrict__ bias, const float* __restrict__ gamma,
             const float* __restrict__ beta, float* __restrict__ Out, int n) {
  __shared__ _Float16 Ws[D * D];
  {
    const uint4* src = (const uint4*)Wsw;
    uint4* dst = (uint4*)Ws;
    for (int i = threadIdx.x; i < (D * D * 2) / 16; i += 256) dst[i] = src[i];
  }
  __syncthreads();
  int wave = threadIdx.x >> 5, lane = threadIdx.x & 31;
  int rowBase = blockIdx.x * 128 + wave * 16;
  int arow = rowBase + (lane & 15);
  int koff = (lane >> 4) * 8;

  v16h a[4];
#pragma unroll
  for (int kt = 0; kt < 4; kt++) a[kt] = load_afrag(A, arow, kt, koff);

  int colLo = lane & 15;
  int mBase = rowBase + ((lane >> 4) << 3);

  v8f acc[8];
#pragma unroll
  for (int nt = 0; nt < 8; nt++) {
    v8f c = {};
#pragma unroll
    for (int kt = 0; kt < 4; kt++) {
      v16h b = *(const v16h*)&Ws[((nt * 4 + kt) * 32 + lane) * 16];
      c = __builtin_amdgcn_wmma_f32_16x16x32_f16(false, a[kt], false, b,
                                                 (short)0, c, false, false);
    }
    float bb = bias[nt * 16 + colLo];
#pragma unroll
    for (int r = 0; r < 8; r++) c[r] += bb;
    acc[nt] = c;
  }

  // LayerNorm row stats. C/D layout: row M = r (+8 for upper half) is striped
  // across one 16-lane half; shfl_xor masks {1,2,4,8} stay within each half.
  float mu[8], rs[8];
#pragma unroll
  for (int r = 0; r < 8; r++) {
    float s = 0.f, q = 0.f;
#pragma unroll
    for (int nt = 0; nt < 8; nt++) { float v = acc[nt][r]; s += v; q += v * v; }
#pragma unroll
    for (int m2 = 1; m2 <= 8; m2 <<= 1) {
      s += __shfl_xor(s, m2, 32);
      q += __shfl_xor(q, m2, 32);
    }
    float mean = s * (1.f / 128.f);
    float var  = q * (1.f / 128.f) - mean * mean;
    mu[r] = mean;
    rs[r] = rsqrtf(var + 1e-5f);
  }

  if (rowBase + 16 <= n) {
    // fast path: wave-uniform, fully in-bounds -> unconditional stores
#pragma unroll
    for (int nt = 0; nt < 8; nt++) {
      int col = nt * 16 + colLo;
      float g = gamma[col], bt = beta[col];
#pragma unroll
      for (int r = 0; r < 8; r++)
        Out[(long long)(mBase + r) * D + col] =
            g * (acc[nt][r] - mu[r]) * rs[r] + bt;
    }
  } else {
    // tail waves of the last block only
#pragma unroll
    for (int nt = 0; nt < 8; nt++) {
      int col = nt * 16 + colLo;
      float g = gamma[col], bt = beta[col];
#pragma unroll
      for (int r = 0; r < 8; r++) {
        int m = mBase + r;
        if (m < n)
          Out[(long long)m * D + col] = g * (acc[nt][r] - mu[r]) * rs[r] + bt;
      }
    }
  }
}

// ---------------------------------------------------------------------------
extern "C" void kernel_launch(void* const* d_in, const int* in_sizes, int n_in,
                              void* d_out, int out_size, void* d_ws, size_t ws_size,
                              hipStream_t stream) {
  const float* x     = (const float*)d_in[0];
  const int*   ei    = (const int*)  d_in[1];   // [2][E] (src row, tgt row)
  const float* ewt   = (const float*)d_in[2];
  const float* W1    = (const float*)d_in[3];
  const float* b1    = (const float*)d_in[4];
  const float* W2    = (const float*)d_in[5];
  const float* b2    = (const float*)d_in[6];
  const float* gamma = (const float*)d_in[7];
  const float* beta  = (const float*)d_in[8];
  float* out = (float*)d_out;

  const int n    = in_sizes[0] / D;          // 100000
  const int E    = in_sizes[2];              // 1600000
  const int nPad = ((n + 127) / 128) * 128;  // padded rows for guard-free GEMM

  // workspace carve-up (256B aligned)
  auto al = [](size_t v) { return (v + 255) & ~(size_t)255; };
  char* w = (char*)d_ws;
  size_t off = 0;
  unsigned* mo = (unsigned*)(w + off); off = al(off + (size_t)n * 4);
  float* dn    = (float*)(w + off);    off = al(off + (size_t)n * 4);
  float* ew    = (float*)(w + off);    off = al(off + (size_t)E * 4);
  float* agg   = (float*)(w + off);    off = al(off + (size_t)n * D * 4);
  _Float16* cmb = (_Float16*)(w + off); off = al(off + (size_t)nPad * D * 2);
  _Float16* h1  = (_Float16*)(w + off); off = al(off + (size_t)nPad * D * 2);
  _Float16* W1s = (_Float16*)(w + off); off = al(off + (size_t)D * D * 2);
  _Float16* W2s = (_Float16*)(w + off); off = al(off + (size_t)D * D * 2);

  long long tot = (long long)n * D;
  k_init   <<<(unsigned)((tot + 255) / 256), 256, 0, stream>>>(agg, mo, dn, tot, n);
  k_wswz   <<<128, 256, 0, stream>>>(W1, W2, W1s, W2s);
  k_emax   <<<(E + 255) / 256, 256, 0, stream>>>(ei, ewt, mo, E);
  k_eexp   <<<(E + 255) / 256, 256, 0, stream>>>(ei, ewt, mo, ew, dn, E);
  k_scatter<<<(E + 7) / 8,     256, 0, stream>>>(ei, ew, dn, x, agg, E);
  k_combine<<<(n + 7) / 8,     256, 0, stream>>>(agg, x, cmb, n);
  k_gemm1  <<<nPad / 128, 256, 0, stream>>>(cmb, W1s, b1, h1);
  k_gemm2  <<<nPad / 128, 256, 0, stream>>>(h1, W2s, b2, gamma, beta, out, n);
}
